// GCN_34720515620910
// MI455X (gfx1250) — matrix operand order repaired and checked by reading, not compiled, not run on hardware
//
#include <hip/hip_runtime.h>
#include <math.h>

typedef __bf16 bf16_t;
typedef __attribute__((ext_vector_type(16))) __bf16 v16bf;
typedef __attribute__((ext_vector_type(8)))  float  v8f;

#define HDIM 128

// ---------------- utility fills ----------------
__global__ void fill_u32(unsigned int* p, unsigned int v, int n) {
    int i = blockIdx.x * blockDim.x + threadIdx.x;
    if (i < n) p[i] = v;
}
// vectorized: n4 = count of float4 elements
__global__ void fill_f32x4(float4* p, float v, int n4) {
    int i = blockIdx.x * blockDim.x + threadIdx.x;
    if (i < n4) p[i] = make_float4(v, v, v, v);
}

// ---------------- degree computation ----------------
__global__ void degree_kernel(const int* __restrict__ src, const int* __restrict__ dst,
                              unsigned int* __restrict__ degOut, unsigned int* __restrict__ degIn,
                              int E) {
    int e = blockIdx.x * blockDim.x + threadIdx.x;
    if (e < E) {
        atomicAdd(&degOut[src[e]], 1u);
        atomicAdd(&degIn[dst[e]], 1u);
    }
}

__global__ void dinv_kernel(const unsigned int* __restrict__ degOut,
                            const unsigned int* __restrict__ degIn,
                            float* __restrict__ dinvOut, float* __restrict__ dinvIn, int N) {
    int i = blockIdx.x * blockDim.x + threadIdx.x;
    if (i < N) {
        unsigned int a = degOut[i]; if (a < 1u) a = 1u;
        unsigned int b = degIn[i];  if (b < 1u) b = 1u;
        dinvOut[i] = rsqrtf((float)a);
        dinvIn[i]  = rsqrtf((float)b);
    }
}

// ---------------- weight transpose + bf16 convert ----------------
// W is [128,128] row-major (k-major). Wt[n*128+k] = (bf16)W[k*128+n].
__global__ void convert_w_kernel(const float* __restrict__ W, bf16_t* __restrict__ Wt) {
    int i = blockIdx.x * blockDim.x + threadIdx.x;   // 16384 threads
    int n = i >> 7, k = i & 127;
    Wt[n * HDIM + k] = (bf16_t)W[k * HDIM + n];
}

// ---------------- WMMA GEMM: out[m,:] = (A[m,:] @ W) * rowScale[m] ----------------
// A: [M,128] f32 row-major. Wt: [128,128] bf16, n-major (Wt[n][k]).
// Block = 256 threads (8 waves). Each block computes 128 rows; wave w -> rows r0..r0+15.
__global__ __launch_bounds__(256)
void gemm_wmma_kernel(const float* __restrict__ A, const bf16_t* __restrict__ Wt,
                      const float* __restrict__ rowScale, float* __restrict__ out, int M) {
    __shared__ __attribute__((aligned(32))) bf16_t WtLds[HDIM * HDIM];  // 32 KB

    // stage transposed bf16 weights into LDS (16B vector copies)
    {
        const uint4* s = (const uint4*)Wt;
        uint4* d = (uint4*)WtLds;
        #pragma unroll
        for (int i = threadIdx.x; i < (HDIM * HDIM) / 8; i += 256) d[i] = s[i];
    }
    __syncthreads();

    const int wave = threadIdx.x >> 5;
    const int lane = threadIdx.x & 31;
    const int r0   = blockIdx.x * 128 + wave * 16;
    const int m    = lane & 15;
    const int half = lane >> 4;

    int rowA = r0 + m;
    if (rowA >= M) rowA = M - 1;               // clamp; stores are guarded below
    const float* ap = A + (size_t)rowA * HDIM;

    // A fragments for all four K-blocks (ISA 16-bit A 16x32 layout):
    // lanes 0-15 hold K = {k0..k0+7, k0+16..k0+23}; lanes 16-31 hold K = {+8, +24}.
    v16bf afrag[4];
    #pragma unroll
    for (int kt = 0; kt < 4; ++kt) {
        const int kb = kt * 32 + half * 8;
        float4 f0 = *(const float4*)(ap + kb + 0);
        float4 f1 = *(const float4*)(ap + kb + 4);
        float4 f2 = *(const float4*)(ap + kb + 16);
        float4 f3 = *(const float4*)(ap + kb + 20);
        v16bf a;
        a[0]  = (bf16_t)f0.x; a[1]  = (bf16_t)f0.y; a[2]  = (bf16_t)f0.z; a[3]  = (bf16_t)f0.w;
        a[4]  = (bf16_t)f1.x; a[5]  = (bf16_t)f1.y; a[6]  = (bf16_t)f1.z; a[7]  = (bf16_t)f1.w;
        a[8]  = (bf16_t)f2.x; a[9]  = (bf16_t)f2.y; a[10] = (bf16_t)f2.z; a[11] = (bf16_t)f2.w;
        a[12] = (bf16_t)f3.x; a[13] = (bf16_t)f3.y; a[14] = (bf16_t)f3.z; a[15] = (bf16_t)f3.w;
        afrag[kt] = a;
    }

    // Hoist the per-row output scales (invariant across column tiles).
    // Lane handles rows r0 + half*8 + r for r = 0..7 (the C/D layout rows).
    float scale[8];
    #pragma unroll
    for (int r = 0; r < 8; ++r) {
        int rowg = r0 + r + half * 8;
        if (rowg >= M) rowg = M - 1;
        scale[r] = rowScale[rowg];
    }

    const int n = lane & 15;
    const bool fullTile = (r0 + 16 <= M);      // wave-uniform tail check
    float* outBase = out + (size_t)(r0 + half * 8) * HDIM + n;

    #pragma unroll
    for (int nt = 0; nt < 8; ++nt) {
        const int c0 = nt * 16;
        v8f acc = {};
        #pragma unroll
        for (int kt = 0; kt < 4; ++kt) {
            // B 32x16 bf16 layout (matches A): lane<16 -> K {0..7,16..23}, lane>=16 -> K {8..15,24..31};
            // N = lane&15. Wt is n-major so each 8-K chunk is contiguous (two 16B LDS loads).
            const bf16_t* bp = &WtLds[(c0 + n) * HDIM + kt * 32 + half * 8];
            v16bf b;
            *((uint4*)&b + 0) = *(const uint4*)bp;
            *((uint4*)&b + 1) = *(const uint4*)(bp + 16);
            acc = __builtin_amdgcn_wmma_f32_16x16x32_bf16(
                false, afrag[kt], false, b, (short)0, acc, false, false);
        }
        // C/D layout: VGPR r, lanes 0-15 -> row r, lanes 16-31 -> row r+8; col = c0 + (lane&15)
        if (fullTile) {
            #pragma unroll
            for (int r = 0; r < 8; ++r)
                outBase[(size_t)r * HDIM + c0] = acc[r] * scale[r];
        } else {
            #pragma unroll
            for (int r = 0; r < 8; ++r) {
                const int rowg = r0 + r + half * 8;
                if (rowg < M) out[(size_t)rowg * HDIM + c0 + n] = acc[r] * scale[r];
            }
        }
    }
}

// ---------------- edge scatter-sum: agg[dst] += hW[src] ----------------
// One wave per edge; each lane moves 4 contiguous floats (512B per edge, coalesced).
__global__ __launch_bounds__(256)
void scatter_kernel(const float* __restrict__ hW, const int* __restrict__ src,
                    const int* __restrict__ dst, float* __restrict__ agg, int E) {
    int gtid = blockIdx.x * blockDim.x + threadIdx.x;
    int e = gtid >> 5;
    int lane = gtid & 31;
    if (e >= E) return;
    const int s = src[e], d = dst[e];
    const float4 v = *(const float4*)(hW + (size_t)s * HDIM + lane * 4);
    float* dp = agg + (size_t)d * HDIM + lane * 4;
    atomicAdd(dp + 0, v.x);
    atomicAdd(dp + 1, v.y);
    atomicAdd(dp + 2, v.z);
    atomicAdd(dp + 3, v.w);
}

// ---------------- epilogue: out = relu(agg * dinvIn[row] + b), float4 granularity ----------------
__global__ void epilogue_kernel(const float4* __restrict__ agg, const float* __restrict__ dinvIn,
                                const float4* __restrict__ b, float4* __restrict__ out, int n4) {
    int i = blockIdx.x * blockDim.x + threadIdx.x;    // n4 = N * HDIM / 4
    if (i >= n4) return;
    const int node = i >> 5;                          // 32 float4 per row
    const int f4   = i & 31;
    const float s  = dinvIn[node];
    float4 a = agg[i];
    float4 bb = b[f4];
    float4 r;
    r.x = fmaxf(a.x * s + bb.x, 0.f);
    r.y = fmaxf(a.y * s + bb.y, 0.f);
    r.z = fmaxf(a.z * s + bb.z, 0.f);
    r.w = fmaxf(a.w * s + bb.w, 0.f);
    out[i] = r;
}

// ---------------- edge-pair classifier: sigmoid(concat(h[s],h[d]) @ Wc + bc) ----------------
// One wave per pair. Wc is [256,2] row-major, bc[2].
__global__ __launch_bounds__(256)
void classify_kernel(const float* __restrict__ h, const int* __restrict__ csrc,
                     const int* __restrict__ cdst, const float* __restrict__ Wc,
                     const float* __restrict__ bc, float* __restrict__ out, int EC) {
    int gtid = blockIdx.x * blockDim.x + threadIdx.x;
    int p = gtid >> 5;
    int lane = gtid & 31;
    if (p >= EC) return;
    const int s = csrc[p], d = cdst[p];
    const float4 hs = *(const float4*)(h + (size_t)s * HDIM + lane * 4);
    const float4 hd = *(const float4*)(h + (size_t)d * HDIM + lane * 4);
    float hsv[4] = {hs.x, hs.y, hs.z, hs.w};
    float hdv[4] = {hd.x, hd.y, hd.z, hd.w};
    float s0 = 0.f, s1 = 0.f;
    const int k0 = lane * 4;
    #pragma unroll
    for (int j = 0; j < 4; ++j) {
        const float* wA = Wc + (size_t)(k0 + j) * 2;          // row for h[s] part
        const float* wB = Wc + (size_t)(HDIM + k0 + j) * 2;   // row for h[d] part
        s0 += hsv[j] * wA[0] + hdv[j] * wB[0];
        s1 += hsv[j] * wA[1] + hdv[j] * wB[1];
    }
    #pragma unroll
    for (int off = 16; off > 0; off >>= 1) {
        s0 += __shfl_xor(s0, off, 32);
        s1 += __shfl_xor(s1, off, 32);
    }
    if (lane == 0) {
        out[(size_t)p * 2 + 0] = 1.f / (1.f + expf(-(s0 + bc[0])));
        out[(size_t)p * 2 + 1] = 1.f / (1.f + expf(-(s1 + bc[1])));
    }
}

// ---------------- host-side orchestration ----------------
extern "C" void kernel_launch(void* const* d_in, const int* in_sizes, int n_in,
                              void* d_out, int out_size, void* d_ws, size_t ws_size,
                              hipStream_t stream) {
    const float* feat = (const float*)d_in[0];
    const int*   gsrc = (const int*)d_in[1];
    const int*   gdst = (const int*)d_in[2];
    const int*   csrc = (const int*)d_in[3];
    const int*   cdst = (const int*)d_in[4];
    const float* W1   = (const float*)d_in[5];
    const float* b1   = (const float*)d_in[6];
    const float* W2   = (const float*)d_in[7];
    const float* b2   = (const float*)d_in[8];
    const float* Wc   = (const float*)d_in[9];
    const float* bc   = (const float*)d_in[10];

    const int N  = in_sizes[0] / HDIM;
    const int E  = in_sizes[1];
    const int EC = in_sizes[3];

    // workspace carve-up (256B aligned regions)
    char* ws = (char*)d_ws;
    auto carve = [&](size_t bytes) {
        void* p = (void*)ws;
        ws += (bytes + 255) & ~(size_t)255;
        return p;
    };
    unsigned int* degOut = (unsigned int*)carve((size_t)N * 4);
    unsigned int* degIn  = (unsigned int*)carve((size_t)N * 4);
    float* dinvOut = (float*)carve((size_t)N * 4);
    float* dinvIn  = (float*)carve((size_t)N * 4);
    bf16_t* Wt1 = (bf16_t*)carve((size_t)HDIM * HDIM * 2);
    bf16_t* Wt2 = (bf16_t*)carve((size_t)HDIM * HDIM * 2);
    float* bufA = (float*)carve((size_t)N * HDIM * 4);  // GEMM output (pre-scatter)
    float* bufB = (float*)carve((size_t)N * HDIM * 4);  // aggregation target
    float* bufC = (float*)carve((size_t)N * HDIM * 4);  // layer-1 hidden
    float* hout   = (float*)d_out;                      // [N,128] final node embeddings
    float* clsout = hout + (size_t)N * HDIM;            // [EC,2] sigmoid logits

    const int T = 256;
    const int nNodeElems = N * HDIM;
    const int n4 = nNodeElems / 4;

    // degrees + normalization factors
    fill_u32<<<(N + T - 1) / T, T, 0, stream>>>(degOut, 0u, N);
    fill_u32<<<(N + T - 1) / T, T, 0, stream>>>(degIn, 0u, N);
    degree_kernel<<<(E + T - 1) / T, T, 0, stream>>>(gsrc, gdst, degOut, degIn, E);
    dinv_kernel<<<(N + T - 1) / T, T, 0, stream>>>(degOut, degIn, dinvOut, dinvIn, N);

    // bf16 transposed weights
    convert_w_kernel<<<(HDIM * HDIM) / T, T, 0, stream>>>(W1, Wt1);
    convert_w_kernel<<<(HDIM * HDIM) / T, T, 0, stream>>>(W2, Wt2);

    const int gemmBlocks = (N + 127) / 128;
    const int scatterBlocks = (int)(((size_t)E * 32 + T - 1) / T);

    // ---- layer 1 ----
    fill_f32x4<<<(n4 + T - 1) / T, T, 0, stream>>>((float4*)bufB, 0.f, n4);
    gemm_wmma_kernel<<<gemmBlocks, T, 0, stream>>>(feat, Wt1, dinvOut, bufA, N);
    scatter_kernel<<<scatterBlocks, T, 0, stream>>>(bufA, gsrc, gdst, bufB, E);
    epilogue_kernel<<<(n4 + T - 1) / T, T, 0, stream>>>(
        (const float4*)bufB, dinvIn, (const float4*)b1, (float4*)bufC, n4);

    // ---- layer 2 ----
    fill_f32x4<<<(n4 + T - 1) / T, T, 0, stream>>>((float4*)bufB, 0.f, n4);
    gemm_wmma_kernel<<<gemmBlocks, T, 0, stream>>>(bufC, Wt2, dinvOut, bufA, N);
    scatter_kernel<<<scatterBlocks, T, 0, stream>>>(bufA, gsrc, gdst, bufB, E);
    epilogue_kernel<<<(n4 + T - 1) / T, T, 0, stream>>>(
        (const float4*)bufB, dinvIn, (const float4*)b2, (float4*)hout, n4);

    // ---- edge-pair classification ----
    const int clsBlocks = (int)(((size_t)EC * 32 + T - 1) / T);
    classify_kernel<<<clsBlocks, T, 0, stream>>>(hout, csrc, cdst, Wc, bc, clsout, EC);
}
